// Decoder_61821759259036
// MI455X (gfx1250) — matrix operand order repaired and checked by reading, not compiled, not run on hardware
//
#include <hip/hip_runtime.h>
#include <hip/hip_bf16.h>
#include <math.h>

// ---------------------------------------------------------------------------
// CDNA5 (gfx1250) types for WMMA
// ---------------------------------------------------------------------------
typedef __attribute__((ext_vector_type(16))) __bf16 v16bf;
typedef __attribute__((ext_vector_type(8)))  float  v8f;

#define Bc 32
#define Tc 32
#define Hc 512
#define Ec 512
#define Vc 32000
#define Mc 512

// ---------------------------------------------------------------------------
// pack_b: fp32 weight B[K,N] (row-major) -> CDNA5 16-bit B-matrix WMMA
// fragment layout, tiled (K/32) x (N/16): each lane's v16bf fragment is ONE
// contiguous 32-byte chunk.  Layout (ISA 7.12.2, 16-bit B 32x16):
// lane L -> n = L%16, half = L/16; element e holds K = (e<8?0:16) + half*8 + (e&7).
// grid: (N/16, K/32), block: 32 threads (one wave).
// ---------------------------------------------------------------------------
__global__ void pack_b(const float* __restrict__ B, __bf16* __restrict__ out,
                       int K, int N) {
  const int nt   = blockIdx.x;
  const int kc   = blockIdx.y;
  const int lane = threadIdx.x;
  const int half = lane >> 4;
  const int n    = lane & 15;
  const int tiles_n = N >> 4;
  __bf16* dst = out + (((size_t)kc * tiles_n + nt) * 32 + lane) * 16;
  const int col = nt * 16 + n;
#pragma unroll
  for (int e = 0; e < 16; ++e) {
    const int k = kc * 32 + ((e & 8) << 1) + half * 8 + (e & 7);
    dst[e] = (__bf16)B[(size_t)k * N + col];
  }
}

// ---------------------------------------------------------------------------
// GEMM core: one wave computes a 16 x 64 strip of C (4 N-tiles), so the A
// fragment (built once per 32-wide K slice, fp32 -> bf16 on the fly) feeds
// FOUR v_wmma_f32_16x16x32_bf16 issues.  A-fragment layout (ISA 7.12.2,
// 16-bit A 16x32): lane m = L%16, half = L/16; element e -> K =
// (e<8?0:16) + half*8 + (e&7).
// Epilogue: bias add; optional maxout over adjacent column pairs (P=2) via
// lane shuffle, writing C[M, N/2].
// ---------------------------------------------------------------------------
__device__ __forceinline__ void gemm4_emit(const v8f& acc,
                                           const float* __restrict__ bias,
                                           float* __restrict__ C,
                                           int N, int domax, int mt, int col,
                                           int half, int mn) {
  const float bv = bias ? bias[col] : 0.0f;
#pragma unroll
  for (int r = 0; r < 8; ++r) {
    const int row = mt * 16 + r + 8 * half;
    const float v = acc[r] + bv;
    if (!domax) {
      C[(size_t)row * N + col] = v;
    } else {
      const float o = __shfl_xor(v, 1, 32);     // partner column (col ^ 1)
      const float w = fmaxf(v, o);
      if ((mn & 1) == 0) C[(size_t)row * (N >> 1) + (col >> 1)] = w;
    }
  }
}

template <typename AF>
__device__ __forceinline__ void gemm4_core(AF getA,
                                           const __bf16* __restrict__ Bp,
                                           const float* __restrict__ bias,
                                           float* __restrict__ C,
                                           int K, int N, int domax,
                                           int mt, int ntq, int lane) {
  const int half = lane >> 4;
  const int mn   = lane & 15;
  const int tiles_n  = N >> 4;
  const int kc_count = K >> 5;
  const size_t kstride = (size_t)tiles_n * 32 * 16;   // packed elems per K slice

  v8f acc0 = {}, acc1 = {}, acc2 = {}, acc3 = {};

  for (int kc = 0; kc < kc_count; ++kc) {
    // ---- A fragment: 8 pairs of K-consecutive fp32 -> bf16 ----
    const float* Ak = getA(kc);
    v16bf a;
#pragma unroll
    for (int j = 0; j < 8; ++j) {
      const int kbase = ((j & 4) << 2) + half * 8 + ((j & 3) << 1);
      a[2 * j]     = (__bf16)Ak[kbase];
      a[2 * j + 1] = (__bf16)Ak[kbase + 1];
    }
    // ---- 4 packed-B fragments: contiguous 32B per lane each ----
    const __bf16* bb = Bp + (((size_t)kc * tiles_n + ntq * 4) * 32 + lane) * 16;
    v16bf b0 = *(const v16bf*)(bb);
    v16bf b1 = *(const v16bf*)(bb + 512);       // +32 lanes * 16 elems
    v16bf b2 = *(const v16bf*)(bb + 1024);
    v16bf b3 = *(const v16bf*)(bb + 1536);
    if (kc + 1 < kc_count)
      __builtin_prefetch(bb + kstride, 0, 3);   // next K slice -> global_prefetch_b8

    acc0 = __builtin_amdgcn_wmma_f32_16x16x32_bf16(false, a, false, b0, (short)0, acc0, false, false);
    acc1 = __builtin_amdgcn_wmma_f32_16x16x32_bf16(false, a, false, b1, (short)0, acc1, false, false);
    acc2 = __builtin_amdgcn_wmma_f32_16x16x32_bf16(false, a, false, b2, (short)0, acc2, false, false);
    acc3 = __builtin_amdgcn_wmma_f32_16x16x32_bf16(false, a, false, b3, (short)0, acc3, false, false);
  }

  const int colbase = ntq * 64 + mn;
  gemm4_emit(acc0, bias, C, N, domax, mt, colbase,      half, mn);
  gemm4_emit(acc1, bias, C, N, domax, mt, colbase + 16, half, mn);
  gemm4_emit(acc2, bias, C, N, domax, mt, colbase + 32, half, mn);
  gemm4_emit(acc3, bias, C, N, domax, mt, colbase + 48, half, mn);
}

// Plain A[M,K] fp32 row-major.  grid = (N/64, M/16), block = 32.
__global__ void wmma_gemm4(const float* __restrict__ A,
                           const __bf16* __restrict__ Bp,
                           const float* __restrict__ bias,
                           float* __restrict__ C,
                           int K, int N, int domax) {
  const int lane = threadIdx.x;
  const int ntq  = blockIdx.x;
  const int mt   = blockIdx.y;
  const int row  = mt * 16 + (lane & 15);
  const float* Arow = A + (size_t)row * K;
  auto getA = [Arow](int kc) { return Arow + kc * 32; };
  gemm4_core(getA, Bp, bias, C, K, N, domax, mt, ntq, lane);
}

// A = concat([s0 (512) | s1 (1024) | s2 (512)]) gathered on the fly
// (segment boundaries are multiples of the 32-wide K slice -> uniform select).
// K must be 2048.  grid = (N/64, M/16), block = 32.
__global__ void wmma_gemm4_cat(const float* __restrict__ s0,
                               const float* __restrict__ s1,
                               const float* __restrict__ s2,
                               const __bf16* __restrict__ Bp,
                               const float* __restrict__ bias,
                               float* __restrict__ C,
                               int K, int N, int domax) {
  const int lane = threadIdx.x;
  const int ntq  = blockIdx.x;
  const int mt   = blockIdx.y;
  const int row  = mt * 16 + (lane & 15);
  const float* r0 = s0 + (size_t)row * 512;
  const float* r1 = s1 + (size_t)row * 1024;
  const float* r2 = s2 + (size_t)row * 512;
  auto getA = [r0, r1, r2](int kc) {
    const int k = kc * 32;
    if (k < 512)  return r0 + k;
    if (k < 1536) return r1 + (k - 512);
    return r2 + (k - 1536);
  };
  gemm4_core(getA, Bp, bias, C, K, N, domax, mt, ntq, lane);
}

// ---------------------------------------------------------------------------
// init: si = 0, emb[b,:] = emb_table[0,:]
// ---------------------------------------------------------------------------
__global__ void init_state(float* __restrict__ si, float* __restrict__ emb,
                           const float* __restrict__ emb_table) {
  const int i = blockIdx.x * blockDim.x + threadIdx.x;  // B*H
  si[i]  = 0.0f;
  emb[i] = emb_table[i & (Ec - 1)];
}

// ---------------------------------------------------------------------------
// attention: per batch row b,
//   e_j   = v_att . tanh(si_proj[b] + enc_proj[b,j])   (b_att folded into enc_proj)
//   alpha = softmax_j(e)
//   ctx   = sum_j alpha_j * enc_out[b,j,:]   (2H = 1024)
// block per b, 256 threads = 8 waves (wave32).
// ---------------------------------------------------------------------------
__global__ void attention(const float* __restrict__ si_proj,
                          const float* __restrict__ enc_proj,
                          const float* __restrict__ v_att,
                          const float* __restrict__ enc_out,
                          float* __restrict__ alphas,   // + t*B*T already
                          float* __restrict__ context) {
  const int b    = blockIdx.x;
  const int tid  = threadIdx.x;
  const int wave = tid >> 5;
  const int lane = tid & 31;
  __shared__ float e_sh[Tc];
  __shared__ float a_sh[Tc];

  const float* sp = si_proj + (size_t)b * Hc;
#pragma unroll
  for (int jj = 0; jj < 4; ++jj) {
    const int j = wave + jj * 8;
    const float* ep = enc_proj + ((size_t)b * Tc + j) * Hc;
    float acc = 0.0f;
    for (int d = lane; d < Hc; d += 32)
      acc += v_att[d] * tanhf(sp[d] + ep[d]);
    for (int off = 16; off > 0; off >>= 1)
      acc += __shfl_down(acc, off, 32);
    if (lane == 0) e_sh[j] = acc;
  }
  __syncthreads();

  if (tid < 32) {
    float e  = e_sh[tid];
    float mx = e;
    for (int off = 16; off > 0; off >>= 1)
      mx = fmaxf(mx, __shfl_xor(mx, off, 32));
    float ex = expf(e - mx);
    float s  = ex;
    for (int off = 16; off > 0; off >>= 1)
      s += __shfl_xor(s, off, 32);
    const float a = ex / s;
    a_sh[tid] = a;
    alphas[b * Tc + tid] = a;
  }
  __syncthreads();

  const float* eb = enc_out + (size_t)b * Tc * (2 * Hc);
  for (int d = tid; d < 2 * Hc; d += 256) {
    float c = 0.0f;
#pragma unroll 4
    for (int j = 0; j < Tc; ++j) c += a_sh[j] * eb[(size_t)j * (2 * Hc) + d];
    context[(size_t)b * (2 * Hc) + d] = c;
  }
}

// ---------------------------------------------------------------------------
// GRU gate math (PyTorch semantics)
// ---------------------------------------------------------------------------
__global__ void gru_gate(const float* __restrict__ gi, const float* __restrict__ gh,
                         const float* __restrict__ si, float* __restrict__ si_new) {
  const int i = blockIdx.x * blockDim.x + threadIdx.x;  // B*H
  const int b = i >> 9;
  const int h = i & (Hc - 1);
  const float* gib = gi + (size_t)b * (3 * Hc);
  const float* ghb = gh + (size_t)b * (3 * Hc);
  const float r = 1.0f / (1.0f + expf(-(gib[h] + ghb[h])));
  const float z = 1.0f / (1.0f + expf(-(gib[Hc + h] + ghb[Hc + h])));
  const float n = tanhf(gib[2 * Hc + h] + r * ghb[2 * Hc + h]);
  si_new[i] = (1.0f - z) * n + z * si[i];
}

// ---------------------------------------------------------------------------
// softmax over V (in place on the d_out slice holding logits), greedy argmax
// (first-index tie-break, matching jnp.argmax) and embedding gather.
// block per b, 256 threads.
// ---------------------------------------------------------------------------
__global__ void softmax_argmax_emb(float* __restrict__ probs,
                                   const float* __restrict__ emb_table,
                                   float* __restrict__ emb_out) {
  const int b   = blockIdx.x;
  const int tid = threadIdx.x;
  float* row = probs + (size_t)b * Vc;
  __shared__ float smax[256];
  __shared__ int   sidx[256];
  __shared__ float ssum[256];

  float mx = -3.402823466e38f;
  int   ax = 0x7fffffff;
  for (int v = tid; v < Vc; v += 256) {
    const float l = row[v];
    if (l > mx) { mx = l; ax = v; }
  }
  smax[tid] = mx; sidx[tid] = ax;
  __syncthreads();
  for (int s = 128; s > 0; s >>= 1) {
    if (tid < s) {
      if (smax[tid + s] > smax[tid] ||
          (smax[tid + s] == smax[tid] && sidx[tid + s] < sidx[tid])) {
        smax[tid] = smax[tid + s];
        sidx[tid] = sidx[tid + s];
      }
    }
    __syncthreads();
  }
  const float gmax = smax[0];
  const int   tok  = sidx[0];

  float s = 0.0f;
  for (int v = tid; v < Vc; v += 256) s += expf(row[v] - gmax);
  ssum[tid] = s;
  __syncthreads();
  for (int st = 128; st > 0; st >>= 1) {
    if (tid < st) ssum[tid] += ssum[tid + st];
    __syncthreads();
  }
  const float inv = 1.0f / ssum[0];
  for (int v = tid; v < Vc; v += 256) row[v] = expf(row[v] - gmax) * inv;

  for (int d = tid; d < Ec; d += 256)
    emb_out[(size_t)b * Ec + d] = emb_table[(size_t)tok * Ec + d];
}

// ---------------------------------------------------------------------------
// host side
// ---------------------------------------------------------------------------
extern "C" void kernel_launch(void* const* d_in, const int* in_sizes, int n_in,
                              void* d_out, int out_size, void* d_ws, size_t ws_size,
                              hipStream_t stream) {
  (void)in_sizes; (void)n_in; (void)out_size; (void)ws_size;

  const float* enc_out   = (const float*)d_in[0];   // [B,T,2H]
  const float* emb_table = (const float*)d_in[1];   // [V,E]
  const float* W_att     = (const float*)d_in[2];   // [3H,H]
  const float* b_att     = (const float*)d_in[3];   // [H]
  const float* v_att     = (const float*)d_in[4];   // [H]
  const float* W_ih      = (const float*)d_in[5];   // [3H+E, 3H]
  const float* W_hh      = (const float*)d_in[6];   // [H, 3H]
  const float* b_ih      = (const float*)d_in[7];
  const float* b_hh      = (const float*)d_in[8];
  const float* W_max     = (const float*)d_in[9];   // [3H+E, 2M]
  const float* b_max     = (const float*)d_in[10];
  const float* W_fc      = (const float*)d_in[11];  // [M, V]
  const float* b_fc      = (const float*)d_in[12];

  // ---- workspace carve-up ----
  char* ws = (char*)d_ws;
  size_t off = 0;
  auto take = [&](size_t bytes) -> char* {
    char* p = ws + off;
    off = (off + bytes + 255) & ~(size_t)255;
    return p;
  };
  __bf16* pWsi  = (__bf16*)take((size_t)Hc * Hc * 2);             // W_att[:H]
  __bf16* pWenc = (__bf16*)take((size_t)(2 * Hc) * Hc * 2);       // W_att[H:]
  __bf16* pWih  = (__bf16*)take((size_t)2048 * 1536 * 2);
  __bf16* pWhh  = (__bf16*)take((size_t)Hc * 1536 * 2);
  __bf16* pWmax = (__bf16*)take((size_t)2048 * 1024 * 2);
  __bf16* pWfc  = (__bf16*)take((size_t)Mc * Vc * 2);
  float* enc_proj = (float*)take((size_t)Bc * Tc * Hc * 4);       // + b_att
  float* siA     = (float*)take((size_t)Bc * Hc * 4);
  float* siB     = (float*)take((size_t)Bc * Hc * 4);
  float* emb     = (float*)take((size_t)Bc * Ec * 4);
  float* si_proj = (float*)take((size_t)Bc * Hc * 4);
  float* context = (float*)take((size_t)Bc * 2 * Hc * 4);
  float* gi      = (float*)take((size_t)Bc * 1536 * 4);
  float* gh      = (float*)take((size_t)Bc * 1536 * 4);
  float* mbuf    = (float*)take((size_t)Bc * Mc * 4);

  float* out_probs  = (float*)d_out;                         // [T,B,V]
  float* out_alphas = (float*)d_out + (size_t)Tc * Bc * Vc;  // [T,B,T]

  // ---- one-time per call: pack weights to bf16 WMMA fragment layout ----
  pack_b<<<dim3(Hc / 16, Hc / 32),       32, 0, stream>>>(W_att,           pWsi,  Hc,     Hc);
  pack_b<<<dim3(Hc / 16, (2 * Hc) / 32), 32, 0, stream>>>(W_att + Hc * Hc, pWenc, 2 * Hc, Hc);
  pack_b<<<dim3(1536 / 16, 2048 / 32),   32, 0, stream>>>(W_ih,            pWih,  2048,   1536);
  pack_b<<<dim3(1536 / 16, Hc / 32),     32, 0, stream>>>(W_hh,            pWhh,  Hc,     1536);
  pack_b<<<dim3(1024 / 16, 2048 / 32),   32, 0, stream>>>(W_max,           pWmax, 2048,   1024);
  pack_b<<<dim3(Vc / 16, Mc / 32),       32, 0, stream>>>(W_fc,            pWfc,  Mc,     Vc);

  // ---- hoist time-invariant attention projection:
  //      enc_proj[b*T+j,:] = enc_out[b,j,:] @ W_att[H:,:] + b_att ----
  wmma_gemm4<<<dim3(Hc / 64, (Bc * Tc) / 16), 32, 0, stream>>>(
      enc_out, pWenc, b_att, enc_proj, 2 * Hc, Hc, 0);

  init_state<<<(Bc * Hc) / 256, 256, 0, stream>>>(siA, emb, emb_table);

  float* si     = siA;
  float* si_nxt = siB;
  for (int t = 0; t < Tc; ++t) {
    // si-dependent half of attention pre-activation (tiny per-step GEMM)
    wmma_gemm4<<<dim3(Hc / 64, Bc / 16), 32, 0, stream>>>(
        si, pWsi, nullptr, si_proj, Hc, Hc, 0);

    attention<<<Bc, 256, 0, stream>>>(
        si_proj, enc_proj, v_att, enc_out,
        out_alphas + (size_t)t * Bc * Tc, context);

    // gi = [si | context | emb] @ W_ih + b_ih   (concat fused into A-gather)
    wmma_gemm4_cat<<<dim3(1536 / 64, Bc / 16), 32, 0, stream>>>(
        si, context, emb, pWih, b_ih, gi, 2048, 1536, 0);
    // gh = si @ W_hh + b_hh
    wmma_gemm4<<<dim3(1536 / 64, Bc / 16), 32, 0, stream>>>(
        si, pWhh, b_hh, gh, Hc, 1536, 0);

    gru_gate<<<(Bc * Hc) / 256, 256, 0, stream>>>(gi, gh, si, si_nxt);

    // m = maxout2([si_new | context | emb] @ W_max + b_max)  (concat + maxout fused)
    wmma_gemm4_cat<<<dim3(1024 / 64, Bc / 16), 32, 0, stream>>>(
        si_nxt, context, emb, pWmax, b_max, mbuf, 2048, 1024, 1);

    // vocab projection straight into the output slice, then in-place softmax
    float* logits = out_probs + (size_t)t * Bc * Vc;
    wmma_gemm4<<<dim3(Vc / 64, Bc / 16), 32, 0, stream>>>(
        mbuf, pWfc, b_fc, logits, Mc, Vc, 0);

    softmax_argmax_emb<<<Bc, 256, 0, stream>>>(logits, emb_table, emb);

    float* tmp = si; si = si_nxt; si_nxt = tmp;
  }
}
